// QuantLLaMaAttention_7378753814643
// MI455X (gfx1250) — matrix operand order repaired and checked by reading, not compiled, not run on hardware
//
#include <hip/hip_runtime.h>
#include <math.h>

typedef _Float16 v16h __attribute__((ext_vector_type(16)));
typedef _Float16 h8   __attribute__((ext_vector_type(8)));
typedef float    v8f  __attribute__((ext_vector_type(8)));

__device__ __forceinline__ v8f wmma16x16x32(v16h a, v16h b, v8f c) {
  // (neg_a, A, neg_b, B, c_mod, C, reuse_a, reuse_b)
  return __builtin_amdgcn_wmma_f32_16x16x32_f16(false, a, false, b, (short)0, c, false, false);
}

// ASYNCcnt-tracked DMA of 16B per lane from global into LDS (no VGPR data path).
__device__ __forceinline__ void async_copy_b128(unsigned int lds_off, const _Float16* g) {
  asm volatile("global_load_async_to_lds_b128 %0, %1, off"
               :: "v"(lds_off), "v"(g) : "memory");
}
// Generic->LDS offset: low 32 bits of a shared-aperture generic address are the
// wave-relative LDS byte offset (ISA 10.2 aperture mapping).
__device__ __forceinline__ unsigned int lds_off(const void* p) {
  return (unsigned int)(uintptr_t)p;
}

// ---------------------------------------------------------------------------
// One-time f32 -> f16 conversion (memory bound).
// ---------------------------------------------------------------------------
__global__ __launch_bounds__(256)
void f32_to_f16_kernel(const float* __restrict__ in, _Float16* __restrict__ out, int n8) {
  const int i = blockIdx.x * 256 + threadIdx.x;
  if (i >= n8) return;
  const float4* p = (const float4*)in + (size_t)i * 2;
  float4 a = p[0], b = p[1];
  h8 h;
  h[0]=(_Float16)a.x; h[1]=(_Float16)a.y; h[2]=(_Float16)a.z; h[3]=(_Float16)a.w;
  h[4]=(_Float16)b.x; h[5]=(_Float16)b.y; h[6]=(_Float16)b.z; h[7]=(_Float16)b.w;
  *(h8*)(out + (size_t)i * 8) = h;
}

// ---------------------------------------------------------------------------
// WMMA GEMM on f16 operands: C[M,N](f32) = A[M,K] * W[N,K]^T
// Block tile 128x128, 256 threads = 8 waves; wave (wm,wn) does 64x32 as 4x2
// fragments. K stepped by 32 with DOUBLE-BUFFERED async-to-LDS staging:
// each thread issues 4 global_load_async_to_lds_b128 per tile; s_wait_asynccnt 4
// retires the previous tile (async loads complete in order).
// ---------------------------------------------------------------------------
#define LSTR 48   // 32 halves + pad; rows = 96B (16B aligned, conflict-spread)

__global__ __launch_bounds__(256)
void gemm_h_wmma(const _Float16* __restrict__ A, const _Float16* __restrict__ W,
                 float* __restrict__ C, int M, int N, int K) {
  __shared__ __align__(16) _Float16 lA[2][128 * LSTR];
  __shared__ __align__(16) _Float16 lB[2][128 * LSTR];

  const int tid  = threadIdx.x;
  const int lane = tid & 31;
  const int wave = tid >> 5;
  const int wm   = wave >> 2;       // 0..1
  const int wn   = wave & 3;        // 0..3
  const int ln   = lane & 15;
  const int hi   = lane >> 4;
  const int m0   = blockIdx.y * 128;
  const int n0   = blockIdx.x * 128;

  const v8f zero8 = {0.f,0.f,0.f,0.f,0.f,0.f,0.f,0.f};
  v8f acc[4][2];
#pragma unroll
  for (int i = 0; i < 4; ++i)
#pragma unroll
    for (int j = 0; j < 2; ++j) acc[i][j] = zero8;

  // 512 16B-chunks per 128x32 tile; thread covers chunks {tid, tid+256}
  auto issue_tile = [&](int k, int buf) {
#pragma unroll
    for (int c = 0; c < 2; ++c) {
      const int ch  = tid + c * 256;
      const int row = ch >> 2;
      const int col = (ch & 3) * 8;
      async_copy_b128(lds_off(&lA[buf][row * LSTR + col]),
                      A + (size_t)(m0 + row) * K + k + col);
      async_copy_b128(lds_off(&lB[buf][row * LSTR + col]),
                      W + (size_t)(n0 + row) * K + k + col);
    }
  };

  const int nk = K >> 5;
  issue_tile(0, 0);

  for (int i = 0; i < nk; ++i) {
    const int cur = i & 1;
    if (i + 1 < nk) {
      issue_tile((i + 1) << 5, cur ^ 1);
      asm volatile("s_wait_asynccnt 4" ::: "memory");  // older tile's 4 done
    } else {
      asm volatile("s_wait_asynccnt 0" ::: "memory");
    }
    __syncthreads();

    // A fragments (16x32): lane row m, runs K = hi*8+j and 16+hi*8+j
    v16h af[4];
#pragma unroll
    for (int fm = 0; fm < 4; ++fm) {
      const _Float16* p = &lA[cur][(wm*64 + fm*16 + ln) * LSTR];
      ((h8*)&af[fm])[0] = *(const h8*)(p + hi*8);
      ((h8*)&af[fm])[1] = *(const h8*)(p + 16 + hi*8);
    }
    // B fragments (32x16): lane col n=ln, runs K = hi*8+j and 16+hi*8+j
    v16h bf[2];
#pragma unroll
    for (int fn = 0; fn < 2; ++fn) {
      const _Float16* p = &lB[cur][(wn*32 + fn*16 + ln) * LSTR];
      ((h8*)&bf[fn])[0] = *(const h8*)(p + hi*8);
      ((h8*)&bf[fn])[1] = *(const h8*)(p + 16 + hi*8);
    }
#pragma unroll
    for (int fm = 0; fm < 4; ++fm)
#pragma unroll
      for (int fn = 0; fn < 2; ++fn)
        acc[fm][fn] = wmma16x16x32(af[fm], bf[fn], acc[fm][fn]);
    __syncthreads();
  }

  // C/D layout: element r -> (M = r + hi*8, N = ln)
#pragma unroll
  for (int fm = 0; fm < 4; ++fm)
#pragma unroll
    for (int fn = 0; fn < 2; ++fn)
#pragma unroll
      for (int r = 0; r < 8; ++r) {
        int row = m0 + wm*64 + fm*16 + r + hi*8;
        int col = n0 + wn*32 + fn*16 + ln;
        C[(size_t)row * N + col] = acc[fm][fn][r];
      }
}

// ---------------------------------------------------------------------------
// RoPE (q,k) + uint4 group quant-dequant (k,v), f32 in -> f16 out.
// One block per (s, h); 128 threads = D.
// ---------------------------------------------------------------------------
__global__ __launch_bounds__(128)
void rope_quant_kernel(const float* __restrict__ qf, const float* __restrict__ kf,
                       const float* __restrict__ vf, const int* __restrict__ pos_ids,
                       const float* __restrict__ ksc, const int* __restrict__ kzp,
                       const float* __restrict__ vsc, const int* __restrict__ vzp,
                       _Float16* __restrict__ qh, _Float16* __restrict__ kh,
                       _Float16* __restrict__ vh, int HID) {
  const int blk = blockIdx.x;
  const int s = blk >> 5;          // H == 32
  const int h = blk & 31;
  const int d = threadIdx.x;       // 0..127
  const size_t base = (size_t)s * HID + h * 128;

  const float pos = (float)pos_ids[s];
  const int fi = d & 63;
  const float inv_freq = __expf(-((float)fi * (1.0f/64.0f)) * 9.210340371976184f); // ln(1e4)
  const float th = pos * inv_freq;
  float cs = cosf(th), sn = sinf(th);

  float q  = qf[base + d];
  float k  = kf[base + d];
  float qp = qf[base + (d ^ 64)];
  float kp = kf[base + (d ^ 64)];
  float rq = (d < 64) ? -qp : qp;
  float rk = (d < 64) ? -kp : kp;
  float qr = q * cs + rq * sn;
  float kr = k * cs + rk * sn;

  const int pi = h * 2 + (d >> 6);     // (head, group) param index, G=64
  {
    float sc = ksc[pi], zp = (float)kzp[pi];
    float qv = fminf(fmaxf(rintf(kr / sc) + zp, 0.f), 15.f);
    kr = (qv - zp) * sc;
  }
  float v = vf[base + d];
  {
    float sc = vsc[pi], zp = (float)vzp[pi];
    float qv = fminf(fmaxf(rintf(v / sc) + zp, 0.f), 15.f);
    v = (qv - zp) * sc;
  }
  qh[base + d] = (_Float16)qr;
  kh[base + d] = (_Float16)kr;
  vh[base + d] = (_Float16)v;
}

// ---------------------------------------------------------------------------
// Causal flash attention: block = (64-query tile, head), 4 waves x 16 rows.
// 32-key tiles: K staged row-major via async-to-LDS, V staged transposed via
// VALU (needs the shuffle anyway); P redistributed through per-wave LDS.
// Output written as f16 so the Wo projection runs on the f16 WMMA GEMM.
// ---------------------------------------------------------------------------
#define KSTR 136   // 128 + 8 pad (row = 272B, 16B aligned)
#define VSTR 40    // 32 + 8 pad
#define PSTR 40

__global__ __launch_bounds__(128)
void flash_attn_wmma(const _Float16* __restrict__ qh, const _Float16* __restrict__ kh,
                     const _Float16* __restrict__ vh, _Float16* __restrict__ o, int HID) {
  __shared__ __align__(16) _Float16 Kt[32 * KSTR];
  __shared__ __align__(16) _Float16 Vt[128 * VSTR];
  __shared__ __align__(16) _Float16 P[4][16 * PSTR];

  const int qb   = blockIdx.x;
  const int h    = blockIdx.y;
  const int tid  = threadIdx.x;
  const int lane = tid & 31;
  const int wave = tid >> 5;
  const int ln   = lane & 15;
  const int hi   = lane >> 4;
  const int qr0  = qb * 64 + wave * 16;
  const int hoff = h * 128;

  // Q fragments (16x32 d-chunks) straight from global
  v16h qfr[4];
  {
    const _Float16* qrow = qh + (size_t)(qr0 + ln) * HID + hoff;
#pragma unroll
    for (int d = 0; d < 4; ++d) {
      ((h8*)&qfr[d])[0] = *(const h8*)(qrow + d*32 + hi*8);
      ((h8*)&qfr[d])[1] = *(const h8*)(qrow + d*32 + 16 + hi*8);
    }
  }

  const v8f zero8 = {0.f,0.f,0.f,0.f,0.f,0.f,0.f,0.f};
  v8f O[8];
#pragma unroll
  for (int c = 0; c < 8; ++c) O[c] = zero8;
  float mrow[8], lrow[8];
#pragma unroll
  for (int r = 0; r < 8; ++r) { mrow[r] = -1e30f; lrow[r] = 0.f; }

  const float scale = 0.08838834764831845f;   // 1/sqrt(128)
  const int nkt = 2 * qb + 2;

  for (int kt = 0; kt < nkt; ++kt) {
    const int kb = kt * 32;
    __syncthreads();
    {
      // K tile: 32x128 halves = 512 16B chunks, async DMA into LDS
#pragma unroll
      for (int c = 0; c < 4; ++c) {
        const int ch  = tid + c * 128;
        const int key = ch >> 4;
        const int col = (ch & 15) * 8;
        async_copy_b128(lds_off(&Kt[key * KSTR + col]),
                        kh + (size_t)(kb + key) * HID + hoff + col);
      }
      // V tile transposed: Vt[d][key]
      const int key = tid >> 2;
      const int db  = (tid & 3) * 32;
      const _Float16* vs = vh + (size_t)(kb + key) * HID + hoff + db;
#pragma unroll
      for (int i = 0; i < 4; ++i) {
        h8 x = *(const h8*)(vs + i*8);
#pragma unroll
        for (int j = 0; j < 8; ++j) Vt[(db + i*8 + j) * VSTR + key] = x[j];
      }
    }
    asm volatile("s_wait_asynccnt 0" ::: "memory");
    __syncthreads();

    if (kb <= qr0 + 15) {    // wave-uniform skip of fully-masked tiles
      v8f s0 = zero8, s1 = zero8;
#pragma unroll
      for (int d = 0; d < 4; ++d) {
        v16h b0, b1;
        const _Float16* p0 = &Kt[ln        * KSTR + d*32 + hi*8];
        const _Float16* p1 = &Kt[(16 + ln) * KSTR + d*32 + hi*8];
        ((h8*)&b0)[0] = *(const h8*)p0;  ((h8*)&b0)[1] = *(const h8*)(p0 + 16);
        ((h8*)&b1)[0] = *(const h8*)p1;  ((h8*)&b1)[1] = *(const h8*)(p1 + 16);
        s0 = wmma16x16x32(qfr[d], b0, s0);
        s1 = wmma16x16x32(qfr[d], b1, s1);
      }

      _Float16* Pw = &P[wave][0];
#pragma unroll
      for (int r = 0; r < 8; ++r) {
        const int row = qr0 + r + hi*8;
        const int c0  = kb + ln;
        float a0 = (c0      <= row) ? s0[r] * scale : -1e30f;
        float a1 = (c0 + 16 <= row) ? s1[r] * scale : -1e30f;
        float mx = fmaxf(a0, a1);
        mx = fmaxf(mx, __shfl_xor(mx, 1, 32));
        mx = fmaxf(mx, __shfl_xor(mx, 2, 32));
        mx = fmaxf(mx, __shfl_xor(mx, 4, 32));
        mx = fmaxf(mx, __shfl_xor(mx, 8, 32));
        const float mnew = fmaxf(mrow[r], mx);
        const float p0 = __expf(a0 - mnew);
        const float p1 = __expf(a1 - mnew);
        float sum = p0 + p1;
        sum += __shfl_xor(sum, 1, 32);
        sum += __shfl_xor(sum, 2, 32);
        sum += __shfl_xor(sum, 4, 32);
        sum += __shfl_xor(sum, 8, 32);
        const float alpha = __expf(mrow[r] - mnew);
        lrow[r] = lrow[r] * alpha + sum;
        mrow[r] = mnew;
#pragma unroll
        for (int c = 0; c < 8; ++c) O[c][r] *= alpha;
        Pw[(r + hi*8) * PSTR + ln]      = (_Float16)p0;
        Pw[(r + hi*8) * PSTR + 16 + ln] = (_Float16)p1;
      }
      asm volatile("s_wait_dscnt 0" ::: "memory");   // cross-lane LDS RAW in wave

      v16h pf;
      {
        const _Float16* pr = &Pw[ln * PSTR];
        ((h8*)&pf)[0] = *(const h8*)(pr + hi*8);
        ((h8*)&pf)[1] = *(const h8*)(pr + 16 + hi*8);
      }
#pragma unroll
      for (int c = 0; c < 8; ++c) {
        v16h bv;
        const _Float16* vr = &Vt[(c*16 + ln) * VSTR + hi*8];
        ((h8*)&bv)[0] = *(const h8*)vr;  ((h8*)&bv)[1] = *(const h8*)(vr + 16);
        O[c] = wmma16x16x32(pf, bv, O[c]);
      }
    }
  }

#pragma unroll
  for (int r = 0; r < 8; ++r) {
    const float inv = 1.0f / lrow[r];
    const int row = qr0 + r + hi*8;
#pragma unroll
    for (int c = 0; c < 8; ++c)
      o[(size_t)row * HID + hoff + c*16 + ln] = (_Float16)(O[c][r] * inv);
  }
}

// ---------------------------------------------------------------------------
extern "C" void kernel_launch(void* const* d_in, const int* in_sizes, int n_in,
                              void* d_out, int out_size, void* d_ws, size_t ws_size,
                              hipStream_t stream) {
  (void)in_sizes; (void)n_in; (void)out_size; (void)ws_size;
  const int S = 2048, HID = 4096;

  const float* X   = (const float*)d_in[0];
  // d_in[1] = attention_mask (causal; applied analytically in flash kernel)
  const int*   pos = (const int*)d_in[2];
  const float* Wq  = (const float*)d_in[3];
  const float* Wk  = (const float*)d_in[4];
  const float* Wv  = (const float*)d_in[5];
  const float* Wo  = (const float*)d_in[6];
  const float* ksc = (const float*)d_in[7];
  const int*   kzp = (const int*)d_in[8];
  const float* vsc = (const float*)d_in[9];
  const int*   vzp = (const int*)d_in[10];
  float* out = (float*)d_out;

  const size_t SZ  = (size_t)S * HID;      // 8M
  const size_t WSZ = (size_t)HID * HID;    // 16M
  _Float16* Xh    = (_Float16*)d_ws;       // 16MB
  _Float16* Wqh   = Xh  + SZ;              // 32MB each
  _Float16* Wkh   = Wqh + WSZ;
  _Float16* Wvh   = Wkh + WSZ;
  _Float16* Woh   = Wvh + WSZ;
  float*    qf    = (float*)(Woh + WSZ);   // 32MB each (f32)
  float*    kf    = qf + SZ;
  float*    vf    = kf + SZ;
  _Float16* qh    = (_Float16*)(vf + SZ);  // 16MB each
  _Float16* khb   = qh + SZ;
  _Float16* vhb   = khb + SZ;
  _Float16* attnh = vhb + SZ;              // 16MB   (total ~304MB)

  const int n8x = (int)(SZ >> 3), n8w = (int)(WSZ >> 3);
  f32_to_f16_kernel<<<(n8x + 255) / 256, 256, 0, stream>>>(X,  Xh,  n8x);
  f32_to_f16_kernel<<<(n8w + 255) / 256, 256, 0, stream>>>(Wq, Wqh, n8w);
  f32_to_f16_kernel<<<(n8w + 255) / 256, 256, 0, stream>>>(Wk, Wkh, n8w);
  f32_to_f16_kernel<<<(n8w + 255) / 256, 256, 0, stream>>>(Wv, Wvh, n8w);
  f32_to_f16_kernel<<<(n8w + 255) / 256, 256, 0, stream>>>(Wo, Woh, n8w);

  dim3 gemm_grid(HID / 128, S / 128);      // 32 x 16
  gemm_h_wmma<<<gemm_grid, 256, 0, stream>>>(Xh, Wqh, qf, S, HID, HID);
  gemm_h_wmma<<<gemm_grid, 256, 0, stream>>>(Xh, Wkh, kf, S, HID, HID);
  gemm_h_wmma<<<gemm_grid, 256, 0, stream>>>(Xh, Wvh, vf, S, HID, HID);

  rope_quant_kernel<<<S * 32, 128, 0, stream>>>(qf, kf, vf, pos,
                                                ksc, kzp, vsc, vzp,
                                                qh, khb, vhb, HID);

  flash_attn_wmma<<<dim3(S / 64, 32), 128, 0, stream>>>(qh, khb, vhb, attnh, HID);

  gemm_h_wmma<<<gemm_grid, 256, 0, stream>>>(attnh, Woh, out, S, HID, HID);
}
